// SimpleRNN_18837726560287
// MI455X (gfx1250) — matrix operand order repaired and checked by reading, not compiled, not run on hardware
//
#include <hip/hip_runtime.h>
#include <hip/hip_bf16.h>
#include <math.h>

// ---------------------------------------------------------------------------
// Problem constants
// ---------------------------------------------------------------------------
#define B_   16384
#define T_   56
#define F_   56
#define H_   20
#define FC1_ 128
#define NC_  2

#define NPAD 80   // padded gate columns: [r(20) z(20) i_n(20) h_n(20)]
#define K0   96   // layer0 K: 56 (x) + 20 (h) -> pad 96  (3 chunks of 32)
#define K1   64   // layer1 K: 20 (h1) + 20 (h) -> pad 64 (2 chunks of 32)
#define WAVES 4
#define ROWS_PER_BLOCK (WAVES * 16)   // 64 batch rows per block

typedef _Float16 v16h __attribute__((ext_vector_type(16)));
typedef _Float16 v8h  __attribute__((ext_vector_type(8)));
typedef _Float16 v4h  __attribute__((ext_vector_type(4)));
typedef _Float16 v2h  __attribute__((ext_vector_type(2)));
typedef float    v8f  __attribute__((ext_vector_type(8)));
typedef float    v4f  __attribute__((ext_vector_type(4)));

// ---------------------------------------------------------------------------
// WMMA fragment gathers (per CDNA5 ISA 7.12.2 VGPR layouts, wave32)
//  A (16x32 f16): lanes 0-15 -> M=0..15, K = {0..7, 16..23}
//                 lanes16-31 -> M=0..15, K = {8..15, 24..31}
//  B (32x16 f16): lanes 0-15 -> N=0..15, K = {0..7, 16..23}
//                 lanes16-31 -> N=0..15, K = {8..15, 24..31}
//  C/D (16x16 f32): vgpr r -> M = r + 8*(lane>=16), N = lane&15
// ---------------------------------------------------------------------------
__device__ __forceinline__ v16h load_a_frag(const _Float16* Ast, int stride,
                                            int m16, int hi, int kc) {
  const _Float16* base = Ast + m16 * stride + kc * 32;
  v8h lo = *(const v8h*)(base + hi * 8);        // K = kc*32 + (hi?8:0) .. +7
  v8h hv = *(const v8h*)(base + 16 + hi * 8);   // K = kc*32 + 16 + (hi?8:0) ..
  return __builtin_shufflevector(lo, hv, 0,1,2,3,4,5,6,7,8,9,10,11,12,13,14,15);
}

__device__ __forceinline__ v16h load_b_frag(const _Float16* WcT, int stride,
                                            int m16, int hi, int kc, int nt) {
  const _Float16* base = WcT + (nt * 16 + m16) * stride + kc * 32;
  v8h lo = *(const v8h*)(base + hi * 8);
  v8h hv = *(const v8h*)(base + 16 + hi * 8);
  return __builtin_shufflevector(lo, hv, 0,1,2,3,4,5,6,7,8,9,10,11,12,13,14,15);
}

__device__ __forceinline__ float sigmoidf_(float x) {
  return 1.0f / (1.0f + __expf(-x));
}

// ---------------------------------------------------------------------------
// Kernel 0: collapse the two (linear!) FC layers into one 2x1120 matrix.
//   fcW[0..1119]    = row 0 of W2@W1
//   fcW[1120..2239] = row 1 of W2@W1
//   fcW[2240..2241] = combined bias
// ---------------------------------------------------------------------------
__global__ void fc_combine_kernel(const float* __restrict__ W1,   // [128][1120]
                                  const float* __restrict__ b1,   // [128]
                                  const float* __restrict__ W2,   // [2][128]
                                  const float* __restrict__ b2,   // [2]
                                  float* __restrict__ fcW) {
  int k = blockIdx.x * blockDim.x + threadIdx.x;
  if (k < T_ * H_) {
    float c0 = 0.f, c1 = 0.f;
    for (int j = 0; j < FC1_; ++j) {
      float w1 = W1[(size_t)j * (T_ * H_) + k];
      c0 += W2[j] * w1;
      c1 += W2[FC1_ + j] * w1;
    }
    fcW[k] = c0;
    fcW[T_ * H_ + k] = c1;
  }
  if (k < NC_) {
    float acc = b2[k];
    for (int j = 0; j < FC1_; ++j) acc += W2[k * FC1_ + j] * b1[j];
    fcW[2 * T_ * H_ + k] = acc;
  }
}

// ---------------------------------------------------------------------------
// Kernel 1: GRU layer 0.  Each wave owns 16 batch rows; per timestep it runs
// A=[x_t | h] (16xK0 f16) against the combined weight (K0 x NPAD) with
// 5 N-tiles x 3 K-chunks of v_wmma_f32_16x16x32_f16, then fp32 gate math.
// Gate lanes each own 10 CONSECUTIVE hidden units (m = lane>>1,
// j = (lane&1)*10 + i) so h1 stores pack into 5 dword stores per lane.
// ---------------------------------------------------------------------------
__global__ __launch_bounds__(128)
void gru_layer0_kernel(const float* __restrict__ x,     // [B][T][F]
                       const float* __restrict__ Wih,   // [60][56]
                       const float* __restrict__ Whh,   // [60][20]
                       const float* __restrict__ bih,   // [60]
                       const float* __restrict__ bhh,   // [60]
                       _Float16* __restrict__ h1out) {  // [B][T][H] f16
  __shared__ __align__(16) _Float16 WcT[NPAD * K0];     // [n][k] transposed
  __shared__ float bb[NPAD];
  __shared__ __align__(16) _Float16 Ast[WAVES][16 * K0];
  __shared__ float gh[WAVES][16 * NPAD];
  __shared__ float hbuf[WAVES][16 * H_];

  const int tid = threadIdx.x;

  // Build combined weight, transposed [N][K], f16, zero-padded.
  for (int i = tid; i < NPAD * K0; i += blockDim.x) {
    int n = i / K0, k = i % K0;
    float v = 0.f;
    if (n < 40) {                                   // fused r,z gates
      if (k < F_)            v = Wih[n * F_ + k];
      else if (k < F_ + H_)  v = Whh[n * H_ + (k - F_)];
    } else if (n < 60) {                            // i_n: x-path only
      if (k < F_)            v = Wih[n * F_ + k];
    } else {                                        // h_n: h-path only (rows 40..59)
      if (k >= F_ && k < F_ + H_) v = Whh[(n - 20) * H_ + (k - F_)];
    }
    WcT[i] = (_Float16)v;
  }
  for (int n = tid; n < NPAD; n += blockDim.x) {
    float v;
    if (n < 40)      v = bih[n] + bhh[n];
    else if (n < 60) v = bih[n];
    else             v = bhh[n - 20];
    bb[n] = v;
  }

  const int w = tid >> 5, lane = tid & 31;
  for (int i = lane; i < 16 * K0; i += 32) Ast[w][i] = (_Float16)0.f;
  for (int i = lane; i < 16 * H_; i += 32) hbuf[w][i] = 0.f;
  __syncthreads();

  const int bbase = blockIdx.x * ROWS_PER_BLOCK + w * 16;
  const int m16 = lane & 15;
  const int hi  = lane >> 4;
  const int gm  = lane >> 1;             // gate row owned by this lane
  const int gj0 = (lane & 1) * 10;       // first of 10 consecutive hidden units

  for (int t = 0; t < T_; ++t) {
    // Stage x_t tile (16 x 56 f32) as f16 into A columns [0..F).
    // 16*56 floats = 224 float4 = exactly 7 per lane; 16B-aligned loads,
    // packed-f16 8B LDS stores.
#pragma unroll
    for (int ii = 0; ii < 7; ++ii) {
      int idx = lane + 32 * ii;
      int r = idx / 14, c4 = idx % 14;
      v4f xv = *(const v4f*)(x + (size_t)(bbase + r) * (T_ * F_) +
                             (size_t)t * F_ + c4 * 4);
      v4h hx;
      hx[0] = (_Float16)xv[0]; hx[1] = (_Float16)xv[1];
      hx[2] = (_Float16)xv[2]; hx[3] = (_Float16)xv[3];
      *(v4h*)&Ast[w][r * K0 + c4 * 4] = hx;
    }
    // Gate pre-activations via WMMA
#pragma unroll
    for (int nt = 0; nt < 5; ++nt) {
      v8f acc = {};
#pragma unroll
      for (int kc = 0; kc < 3; ++kc) {
        v16h a = load_a_frag(&Ast[w][0], K0, m16, hi, kc);
        v16h b = load_b_frag(WcT, K0, m16, hi, kc, nt);
        acc = __builtin_amdgcn_wmma_f32_16x16x32_f16(
            false, a, false, b, (short)0, acc, false, false);
      }
      int n  = nt * 16 + m16;
      int mb = hi * 8;
#pragma unroll
      for (int r8 = 0; r8 < 8; ++r8) gh[w][(mb + r8) * NPAD + n] = acc[r8];
    }
    // fp32 gate math: lane owns hidden units j = gj0..gj0+9 of row gm
    float hv[10];
#pragma unroll
    for (int i = 0; i < 10; ++i) {
      int j = gj0 + i;
      float rp = gh[w][gm * NPAD + j]      + bb[j];
      float zp = gh[w][gm * NPAD + 20 + j] + bb[20 + j];
      float in = gh[w][gm * NPAD + 40 + j] + bb[40 + j];
      float hn = gh[w][gm * NPAD + 60 + j] + bb[60 + j];
      float rg = sigmoidf_(rp);
      float zg = sigmoidf_(zp);
      float ng = tanhf(in + rg * hn);
      float hold = hbuf[w][gm * H_ + j];
      float hnew = (1.f - zg) * ng + zg * hold;
      hbuf[w][gm * H_ + j] = hnew;
      hv[i] = hnew;
    }
    // packed writeback: A-staging h-region (LDS, 4B) + h1 (global, 4B dwords)
    _Float16* hdst = h1out + (size_t)(bbase + gm) * (T_ * H_) +
                     (size_t)t * H_ + gj0;
#pragma unroll
    for (int q = 0; q < 5; ++q) {
      v2h p; p[0] = (_Float16)hv[2 * q]; p[1] = (_Float16)hv[2 * q + 1];
      *(v2h*)&Ast[w][gm * K0 + F_ + gj0 + 2 * q] = p;
      *(v2h*)(hdst + 2 * q) = p;
    }
  }
}

// ---------------------------------------------------------------------------
// Kernel 2: GRU layer 1 with the collapsed FC head fused in.  Same WMMA
// structure (K=64 -> 2 chunks).  Each lane accumulates y[m][c] contributions
// for its 10 consecutive hidden units across all timesteps; h2 never goes
// to HBM.  Final y reduced per wave via LDS float atomics (ds_add_f32).
// ---------------------------------------------------------------------------
__global__ __launch_bounds__(128)
void gru_layer1_fc_kernel(const _Float16* __restrict__ h1,  // [B][T][H] f16
                          const float* __restrict__ Wih,    // [60][20]
                          const float* __restrict__ Whh,    // [60][20]
                          const float* __restrict__ bih,
                          const float* __restrict__ bhh,
                          const float* __restrict__ fcW,    // [2][1120] + bias[2]
                          float* __restrict__ out) {        // [B][2]
  __shared__ __align__(16) _Float16 WcT[NPAD * K1];
  __shared__ float bb[NPAD];
  __shared__ __align__(16) _Float16 Ast[WAVES][16 * K1];
  __shared__ float gh[WAVES][16 * NPAD];
  __shared__ float hbuf[WAVES][16 * H_];
  __shared__ float yacc[WAVES][32];                 // 16 rows x 2 classes

  const int tid = threadIdx.x;

  for (int i = tid; i < NPAD * K1; i += blockDim.x) {
    int n = i / K1, k = i % K1;
    float v = 0.f;
    if (n < 40) {
      if (k < H_)            v = Wih[n * H_ + k];
      else if (k < 2 * H_)   v = Whh[n * H_ + (k - H_)];
    } else if (n < 60) {
      if (k < H_)            v = Wih[n * H_ + k];
    } else {
      if (k >= H_ && k < 2 * H_) v = Whh[(n - 20) * H_ + (k - H_)];
    }
    WcT[i] = (_Float16)v;
  }
  for (int n = tid; n < NPAD; n += blockDim.x) {
    float v;
    if (n < 40)      v = bih[n] + bhh[n];
    else if (n < 60) v = bih[n];
    else             v = bhh[n - 20];
    bb[n] = v;
  }

  const int w = tid >> 5, lane = tid & 31;
  for (int i = lane; i < 16 * K1; i += 32) Ast[w][i] = (_Float16)0.f;
  for (int i = lane; i < 16 * H_; i += 32) hbuf[w][i] = 0.f;
  if (lane < 32) yacc[w][lane] = 0.f;
  __syncthreads();

  const int bbase = blockIdx.x * ROWS_PER_BLOCK + w * 16;
  const int m16 = lane & 15;
  const int hi  = lane >> 4;
  const int gm  = lane >> 1;
  const int gj0 = (lane & 1) * 10;

  float yl0 = 0.f, yl1 = 0.f;            // per-lane FC partials for row gm

  for (int t = 0; t < T_; ++t) {
    // Stage h1[.,t,:] tile (16 x 20 f16): 80 8-byte chunks, 8B-aligned
    for (int idx = lane; idx < 80; idx += 32) {
      int r = idx / 5, c = idx % 5;
      v4h hx = *(const v4h*)(h1 + (size_t)(bbase + r) * (T_ * H_) +
                             (size_t)t * H_ + c * 4);
      *(v4h*)&Ast[w][r * K1 + c * 4] = hx;
    }
#pragma unroll
    for (int nt = 0; nt < 5; ++nt) {
      v8f acc = {};
#pragma unroll
      for (int kc = 0; kc < 2; ++kc) {
        v16h a = load_a_frag(&Ast[w][0], K1, m16, hi, kc);
        v16h b = load_b_frag(WcT, K1, m16, hi, kc, nt);
        acc = __builtin_amdgcn_wmma_f32_16x16x32_f16(
            false, a, false, b, (short)0, acc, false, false);
      }
      int n  = nt * 16 + m16;
      int mb = hi * 8;
#pragma unroll
      for (int r8 = 0; r8 < 8; ++r8) gh[w][(mb + r8) * NPAD + n] = acc[r8];
    }
    float hv[10];
#pragma unroll
    for (int i = 0; i < 10; ++i) {
      int j = gj0 + i;
      float rp = gh[w][gm * NPAD + j]      + bb[j];
      float zp = gh[w][gm * NPAD + 20 + j] + bb[20 + j];
      float in = gh[w][gm * NPAD + 40 + j] + bb[40 + j];
      float hn = gh[w][gm * NPAD + 60 + j] + bb[60 + j];
      float rg = sigmoidf_(rp);
      float zg = sigmoidf_(zp);
      float ng = tanhf(in + rg * hn);
      float hold = hbuf[w][gm * H_ + j];
      float hnew = (1.f - zg) * ng + zg * hold;
      hbuf[w][gm * H_ + j] = hnew;
      hv[i] = hnew;
      // fused (collapsed) FC head: y[gm][c] += h2 * Wc[c][t*H + j]
      int fk = t * H_ + j;
      yl0 += hnew * fcW[fk];
      yl1 += hnew * fcW[T_ * H_ + fk];
    }
#pragma unroll
    for (int q = 0; q < 5; ++q) {
      v2h p; p[0] = (_Float16)hv[2 * q]; p[1] = (_Float16)hv[2 * q + 1];
      *(v2h*)&Ast[w][gm * K1 + H_ + gj0 + 2 * q] = p;
    }
  }
  // reduce the two half-row partials per row (ds_add_f32)
  atomicAdd(&yacc[w][gm * 2 + 0], yl0);
  atomicAdd(&yacc[w][gm * 2 + 1], yl1);
  {
    int m = lane >> 1, c = lane & 1;
    out[(size_t)(bbase + m) * NC_ + c] =
        yacc[w][lane] + fcW[2 * T_ * H_ + c];
  }
}

// ---------------------------------------------------------------------------
// Launcher
// ---------------------------------------------------------------------------
extern "C" void kernel_launch(void* const* d_in, const int* in_sizes, int n_in,
                              void* d_out, int out_size, void* d_ws,
                              size_t ws_size, hipStream_t stream) {
  const float* x     = (const float*)d_in[0];
  const float* Wih0  = (const float*)d_in[1];
  const float* Whh0  = (const float*)d_in[2];
  const float* bih0  = (const float*)d_in[3];
  const float* bhh0  = (const float*)d_in[4];
  const float* Wih1  = (const float*)d_in[5];
  const float* Whh1  = (const float*)d_in[6];
  const float* bih1  = (const float*)d_in[7];
  const float* bhh1  = (const float*)d_in[8];
  const float* Wfc1  = (const float*)d_in[9];
  const float* bfc1  = (const float*)d_in[10];
  const float* Wfc2  = (const float*)d_in[11];
  const float* bfc2  = (const float*)d_in[12];

  // workspace: [0, 16KB) combined FC weights+bias (f32); then h1 as f16
  float*    fcW = (float*)d_ws;
  _Float16* h1  = (_Float16*)((char*)d_ws + 16384);

  fc_combine_kernel<<<(T_ * H_ + 255) / 256, 256, 0, stream>>>(
      Wfc1, bfc1, Wfc2, bfc2, fcW);

  gru_layer0_kernel<<<B_ / ROWS_PER_BLOCK, WAVES * 32, 0, stream>>>(
      x, Wih0, Whh0, bih0, bhh0, h1);

  gru_layer1_fc_kernel<<<B_ / ROWS_PER_BLOCK, WAVES * 32, 0, stream>>>(
      h1, Wih1, Whh1, bih1, bhh1, fcW, (float*)d_out);
}
